// RNN_LSTM_EMBED_90907277787356
// MI455X (gfx1250) — compile-verified, matrix operand order checked by reading, not run on hardware
//
#include <hip/hip_runtime.h>
#include <hip/hip_bf16.h>
#include <math.h>

typedef __attribute__((ext_vector_type(16))) __bf16 v16bf;
typedef __attribute__((ext_vector_type(8)))  float  v8f;
typedef unsigned short u16;

#define BATCH 64
#define TSTEPS 512
#define EMB 512
#define HID 1024
#define NCLS 256
#define NWG 64
#define SLICE 16   // hidden units per workgroup

__device__ __forceinline__ u16 f2bf(float f) {
    unsigned x = __float_as_uint(f);
    unsigned r = (x + 0x7fffu + ((x >> 16) & 1u)) >> 16;   // round-nearest-even
    return (u16)r;
}

// Load a 16x32 bf16 tile (row-major source, stride in elements) into WMMA
// A/B operand layout: lane L<16 -> row L, K chunks [0..7]+[16..23];
// lane L>=16 -> row L-16, K chunks [8..15]+[24..31]. Two 16B loads per lane.
__device__ __forceinline__ v16bf ldtile(const u16* base, int stride, int lane) {
    int r  = lane & 15;
    int hi = (lane >> 4) & 1;
    const u16* p = base + r * stride + hi * 8;
    union { v16bf v; uint4 q[2]; } u;
    u.q[0] = *(const uint4*)(p);
    u.q[1] = *(const uint4*)(p + 16);
    return u.v;
}

__device__ __forceinline__ v8f wmma_bf16(v16bf a, v16bf b, v8f c) {
    return __builtin_amdgcn_wmma_f32_16x16x32_bf16(false, a, false, b,
                                                   (short)0, c, false, false);
}

// ---------------- prep: embedding gather -> bf16 [T][B][E], init barrier ----
__global__ void prep_embed(const int* __restrict__ x,
                           const float* __restrict__ embed,
                           u16* __restrict__ xe,
                           unsigned* __restrict__ bar) {
    int bid = blockIdx.x;                 // bid = t*64 + b
    if (bid == 0 && threadIdx.x == 0) *bar = 0u;
    int t = bid >> 6, b = bid & 63;
    int cls = x[b * TSTEPS + t];
    const float* src = embed + (size_t)cls * EMB;
    u16* dst = xe + (size_t)bid * EMB;
    for (int e = threadIdx.x; e < EMB; e += 64) dst[e] = f2bf(src[e]);
}

// ---------------- prep: fc_W fp32 -> bf16 ----------------
__global__ void prep_fcw(const float* __restrict__ fcW, u16* __restrict__ fw) {
    int i = blockIdx.x * 256 + threadIdx.x;  // 256*1024 elements, grid 1024
    fw[i] = f2bf(fcW[i]);
}

// ---------------- persistent LSTM recurrence ----------------
// 64 WGs x 128 threads. WG g owns hidden cols [16g,16g+16). Wave w owns gate w.
__global__ __launch_bounds__(128, 1)
void lstm_persistent(const float* __restrict__ Wih, const float* __restrict__ Whh,
                     const float* __restrict__ bih, const float* __restrict__ bhh,
                     const u16* __restrict__ xe, u16* __restrict__ hs,
                     unsigned* __restrict__ bar) {
    __shared__ alignas(16) u16  sWhh[4][SLICE][HID];  // 128 KB, resident all 512 steps
    __shared__ alignas(16) u16  sWih[4][SLICE][EMB];  //  64 KB
    __shared__ alignas(16) float sG[4][BATCH][SLICE]; //  16 KB gate exchange
    __shared__ alignas(16) float sC[BATCH][SLICE];    //   4 KB cell state (never leaves LDS)

    const int wg  = blockIdx.x;
    const int j0  = wg * SLICE;
    const int tid = threadIdx.x;
    const int wave = tid >> 5, lane = tid & 31;
    const int gate = wave;

    // One-time: convert this WG's weight slices fp32 -> bf16 into LDS.
    for (int row = 0; row < 4 * SLICE; ++row) {
        int g = row >> 4, jr = row & 15;
        const float* s1 = Whh + ((size_t)g * HID + j0 + jr) * HID;
        for (int k = tid * 4; k < HID; k += 512) {
            float4 f = *(const float4*)(s1 + k);
            sWhh[g][jr][k+0] = f2bf(f.x); sWhh[g][jr][k+1] = f2bf(f.y);
            sWhh[g][jr][k+2] = f2bf(f.z); sWhh[g][jr][k+3] = f2bf(f.w);
        }
        const float* s2 = Wih + ((size_t)g * HID + j0 + jr) * EMB;
        for (int k = tid * 4; k < EMB; k += 512) {
            float4 f = *(const float4*)(s2 + k);
            sWih[g][jr][k+0] = f2bf(f.x); sWih[g][jr][k+1] = f2bf(f.y);
            sWih[g][jr][k+2] = f2bf(f.z); sWih[g][jr][k+3] = f2bf(f.w);
        }
    }
    for (int i = tid; i < BATCH * SLICE; i += 128) ((float*)sC)[i] = 0.f;

    // Per-thread constants for the elementwise phase (j fixed per thread).
    const int jj = tid & 15;
    const int brow = tid >> 4;          // 0..7
    float bsum[4];
    #pragma unroll
    for (int g = 0; g < 4; ++g)
        bsum[g] = bih[g * HID + j0 + jj] + bhh[g * HID + j0 + jj];

    __syncthreads();

    const v8f zero = {0.f,0.f,0.f,0.f,0.f,0.f,0.f,0.f};

    for (int t = 0; t < TSTEPS; ++t) {
        v8f acc[4] = {zero, zero, zero, zero};     // 4 batch tiles of this gate

        // x-side: A = x_embed[t]  (bf16, [64][512]),  B = W_ih slice in LDS
        const u16* xa = xe + (size_t)t * BATCH * EMB;
        for (int k = 0; k < EMB; k += 32) {
            v16bf bt = ldtile(&sWih[gate][0][k], EMB, lane);
            #pragma unroll
            for (int m = 0; m < 4; ++m) {
                v16bf at = ldtile(xa + (size_t)(m * 16) * EMB + k, EMB, lane);
                acc[m] = wmma_bf16(at, bt, acc[m]);
            }
        }
        // h-side: A = h[t-1] (bf16 global, L2-hot), B = W_hh slice in LDS
        if (t > 0) {
            const u16* ha = hs + (size_t)(t - 1) * BATCH * HID;
            for (int k = 0; k < HID; k += 32) {
                v16bf bt = ldtile(&sWhh[gate][0][k], HID, lane);
                #pragma unroll
                for (int m = 0; m < 4; ++m) {
                    v16bf at = ldtile(ha + (size_t)(m * 16) * HID + k, HID, lane);
                    acc[m] = wmma_bf16(at, bt, acc[m]);
                }
            }
        }

        // Park this wave's gate tile in LDS (C/D layout: lane->col, vgpr->row).
        {
            int col = lane & 15, hi = lane >> 4;
            #pragma unroll
            for (int m = 0; m < 4; ++m)
                #pragma unroll
                for (int i = 0; i < 8; ++i)
                    sG[gate][m * 16 + i + hi * 8][col] = acc[m][i];
        }
        __syncthreads();

        // Elementwise LSTM cell update: 1024 (b,j) pairs / 128 threads.
        #pragma unroll
        for (int q = 0; q < 8; ++q) {
            int b = brow + 8 * q;
            float gi = sG[0][b][jj] + bsum[0];
            float gf = sG[1][b][jj] + bsum[1];
            float gg = sG[2][b][jj] + bsum[2];
            float go = sG[3][b][jj] + bsum[3];
            gi = 1.f / (1.f + __expf(-gi));
            gf = 1.f / (1.f + __expf(-gf));
            gg = tanhf(gg);
            go = 1.f / (1.f + __expf(-go));
            float c = gf * sC[b][jj] + gi * gg;
            sC[b][jj] = c;
            float h = go * tanhf(c);
            hs[(size_t)t * BATCH * HID + (size_t)b * HID + j0 + jj] = f2bf(h);
        }

        // Device-wide step barrier: all 64 WGs must publish h[t] before t+1.
        __threadfence();
        __syncthreads();
        if (tid == 0) {
            atomicAdd(bar, 1u);
            unsigned target = (unsigned)(t + 1) * NWG;
            while (__hip_atomic_load(bar, __ATOMIC_RELAXED,
                                     __HIP_MEMORY_SCOPE_AGENT) < target)
                __builtin_amdgcn_s_sleep(1);
        }
        __syncthreads();
        __threadfence();
    }
}

// ---------------- final FC: out[b][t][c] = hs[t][b][:] . fc_W[c][:] + fc_b ---
// grid 512 x 128: wave handles one M-tile (16 rows of (t,b)) x all 256 classes.
__global__ __launch_bounds__(128)
void fc_gemm(const u16* __restrict__ hs, const u16* __restrict__ fw,
             const float* __restrict__ fcb, float* __restrict__ out) {
    const int wave = threadIdx.x >> 5, lane = threadIdx.x & 31;
    const int mt = blockIdx.x * 4 + wave;       // 2048 M-tiles total
    const int t  = mt >> 2;
    const int mb = (mt & 3) * 16;               // batch base
    const u16* A = hs + (size_t)t * BATCH * HID + (size_t)mb * HID;

    const v8f zero = {0.f,0.f,0.f,0.f,0.f,0.f,0.f,0.f};
    v8f acc[16];
    #pragma unroll
    for (int n = 0; n < 16; ++n) acc[n] = zero;

    for (int k = 0; k < HID; k += 32) {
        v16bf at = ldtile(A + k, HID, lane);
        #pragma unroll
        for (int n = 0; n < 16; ++n) {
            v16bf bt = ldtile(fw + (size_t)(n * 16) * HID + k, HID, lane);
            acc[n] = wmma_bf16(at, bt, acc[n]);
        }
    }

    const int col = lane & 15, hi = lane >> 4;
    #pragma unroll
    for (int n = 0; n < 16; ++n) {
        int c = n * 16 + col;
        float bias = fcb[c];
        #pragma unroll
        for (int i = 0; i < 8; ++i) {
            int b = mb + i + hi * 8;
            out[(size_t)b * (TSTEPS * NCLS) + (size_t)t * NCLS + c] = acc[n][i] + bias;
        }
    }
}

extern "C" void kernel_launch(void* const* d_in, const int* in_sizes, int n_in,
                              void* d_out, int out_size, void* d_ws, size_t ws_size,
                              hipStream_t stream) {
    (void)in_sizes; (void)n_in; (void)out_size; (void)ws_size;
    const int*   x     = (const int*)  d_in[0];
    const float* embed = (const float*)d_in[1];
    const float* Wih   = (const float*)d_in[2];
    const float* Whh   = (const float*)d_in[3];
    const float* bih   = (const float*)d_in[4];
    const float* bhh   = (const float*)d_in[5];
    const float* fcW   = (const float*)d_in[6];
    const float* fcb   = (const float*)d_in[7];
    float* out = (float*)d_out;

    char* ws = (char*)d_ws;
    unsigned* bar = (unsigned*)ws;                                   // 256 B
    u16* xe = (u16*)(ws + 256);                                      // 32 MB
    u16* fw = (u16*)(ws + 256 + (size_t)TSTEPS * BATCH * EMB * 2);   // 512 KB
    u16* hs = (u16*)(ws + 256 + (size_t)TSTEPS * BATCH * EMB * 2
                               + (size_t)NCLS * HID * 2);            // 64 MB

    prep_embed<<<TSTEPS * BATCH, 64, 0, stream>>>(x, embed, xe, bar);
    prep_fcw<<<(NCLS * HID) / 256, 256, 0, stream>>>(fcW, fw);
    lstm_persistent<<<NWG, 128, 0, stream>>>(Wih, Whh, bih, bhh, xe, hs, bar);
    fc_gemm<<<(TSTEPS * BATCH / 16) / 4, 128, 0, stream>>>(hs, fw, fcb, out);
}
